// ContactModel_944892805455
// MI455X (gfx1250) — compile-verified, matrix operand order checked by reading, not run on hardware
//
#include <hip/hip_runtime.h>
#include <hip/hip_bf16.h>

// ContactModel for MI455X (gfx1250, wave32).
//
// Shapes: B=4, L=256, D=256, CB=64, H=4, hd=64, T=L+1=257.
//
// Strategy: fold wk into (wq@wk^T) and wv into (wv@wo) per head so the cross
// attention never materializes K/V over the 67 MB pair tensor; the model is
// then bandwidth-bound on one streaming pass of pair per cross layer
// (2 x 67 MB / 23.3 TB/s ~ 5.8 us floor). Dense projections use a wave32
// WMMA f16 GEMM (v_wmma_f32_16x16x32_f16): K templated (static WMMA chains),
// 16x32 tile per wave (two accumulators, shared A fragment), and weights
// pre-packed to f16 in B-fragment order so each lane's fragment is one
// contiguous 32-byte v16h load (2x global_load_b128) instead of 16 strided
// global_load_b32 + serializing waits.

typedef __attribute__((ext_vector_type(16))) _Float16 v16h;
typedef __attribute__((ext_vector_type(8)))  float    v8f;

#define BB 4
#define LL 256
#define DD 256
#define CBW 64
#define TT 257

// ---------------------------------------------------------------------------
// Block-wide reductions (blockDim.x == 256)
// ---------------------------------------------------------------------------
__device__ __forceinline__ float block_reduce_max(float v, float* red) {
  const int t = threadIdx.x;
  red[t] = v; __syncthreads();
  for (int off = 128; off > 0; off >>= 1) {
    if (t < off) red[t] = fmaxf(red[t], red[t + off]);
    __syncthreads();
  }
  float r = red[0]; __syncthreads();
  return r;
}
__device__ __forceinline__ float block_reduce_sum(float v, float* red) {
  const int t = threadIdx.x;
  red[t] = v; __syncthreads();
  for (int off = 128; off > 0; off >>= 1) {
    if (t < off) red[t] += red[t + off];
    __syncthreads();
  }
  float r = red[0]; __syncthreads();
  return r;
}

// ---------------------------------------------------------------------------
// Weight pre-pack: fp32 W[K,N] -> f16 in B-fragment order.
// For each 32x16 (k,n) block: 512 f16, index = block*512 + lane*16 + e with
//   n = nblk*16 + (lane&15)
//   k = kblk*32 + (e>=8)*16 + (lane>=16)*8 + (e&7)
// One thread per packed element; packed writes are fully coalesced.
// grid = K*N/256, block 256.
// ---------------------------------------------------------------------------
__global__ void wpack_kernel(const float* __restrict__ W,
                             _Float16* __restrict__ Wp, int N) {
  const int tid  = blockIdx.x * 256 + threadIdx.x;
  const int blk  = tid >> 9;
  const int lane = (tid >> 4) & 31;
  const int e    = tid & 15;
  const int nT   = N >> 4;
  const int kblk = blk / nT, nblk = blk % nT;
  const int n = nblk * 16 + (lane & 15);
  const int k = kblk * 32 + ((e >= 8) ? 16 : 0) + ((lane >> 4) ? 8 : 0) + (e & 7);
  Wp[tid] = (_Float16)W[(size_t)k * N + n];
}

// ---------------------------------------------------------------------------
// WMMA GEMM: out[M,N] = act(A[M,K] @ W[K,N] + bias (+ residual))
// A fp32 (converted on the fly, float4 loads), W pre-packed f16 (v16h loads).
// One wave computes a 16x32 tile (two accumulators, shared A fragment);
// 8 waves (256 thr) per block. Fragment layouts per CDNA5 ISA 7.12.2.
// ---------------------------------------------------------------------------
template <int K, bool HAS_RES, bool RELU>
__global__ void gemm_wmma_kernel(const float* __restrict__ A,
                                 const _Float16* __restrict__ Wp,
                                 const float* __restrict__ bias,
                                 const float* __restrict__ res,
                                 float* __restrict__ out,
                                 int M, int N) {
  const int lane  = threadIdx.x & 31;
  const int wave  = threadIdx.x >> 5;
  const int n0    = blockIdx.x * 32;
  const int tileM = blockIdx.y * 8 + wave;
  const int m0    = tileM * 16;
  if (m0 >= M) return;                       // no barriers below: safe early-out
  const int rA  = m0 + (lane & 15);
  const int rAc = (rA < M) ? rA : (M - 1);   // clamp loads for ragged M
  const int cB0 = n0 + (lane & 15);
  const int khi = (lane >> 4) ? 8 : 0;
  const int nT  = N >> 4;
  const float* __restrict__ Arow = A + (size_t)rAc * K;
  // packed base for this wave's two B sub-tiles at kk=0
  const _Float16* __restrict__ Wb = Wp + ((size_t)(n0 >> 4)) * 512 + lane * 16;

  v8f acc0 = {}, acc1 = {};
#pragma unroll 8
  for (int kk = 0; kk < K; kk += 32) {
    // A fragment: two contiguous 8-float runs -> 4x float4 (32B aligned)
    const float4 f0 = *(const float4*)(Arow + kk + khi);
    const float4 f1 = *(const float4*)(Arow + kk + khi + 4);
    const float4 f2 = *(const float4*)(Arow + kk + 16 + khi);
    const float4 f3 = *(const float4*)(Arow + kk + 16 + khi + 4);
    v16h a;
    a[0]  = (_Float16)f0.x; a[1]  = (_Float16)f0.y;
    a[2]  = (_Float16)f0.z; a[3]  = (_Float16)f0.w;
    a[4]  = (_Float16)f1.x; a[5]  = (_Float16)f1.y;
    a[6]  = (_Float16)f1.z; a[7]  = (_Float16)f1.w;
    a[8]  = (_Float16)f2.x; a[9]  = (_Float16)f2.y;
    a[10] = (_Float16)f2.z; a[11] = (_Float16)f2.w;
    a[12] = (_Float16)f3.x; a[13] = (_Float16)f3.y;
    a[14] = (_Float16)f3.z; a[15] = (_Float16)f3.w;
    // B fragments: contiguous 32-byte packed loads
    const _Float16* wb = Wb + (size_t)(kk >> 5) * nT * 512;
    const v16h b0 = *(const v16h*)(wb);
    const v16h b1 = *(const v16h*)(wb + 512);
    acc0 = __builtin_amdgcn_wmma_f32_16x16x32_f16(
        false, a, false, b0, (short)0, acc0, false, false);
    acc1 = __builtin_amdgcn_wmma_f32_16x16x32_f16(
        false, a, false, b1, (short)0, acc1, false, false);
  }

  const int cB1 = cB0 + 16;
  const float bias0 = bias[cB0];
  const float bias1 = bias[cB1];
#pragma unroll
  for (int r = 0; r < 8; ++r) {
    const int row = m0 + r + ((lane >> 4) ? 8 : 0);
    if (row < M) {
      float v0 = acc0[r] + bias0;
      float v1 = acc1[r] + bias1;
      if (HAS_RES) {
        v0 += res[(size_t)row * N + cB0];
        v1 += res[(size_t)row * N + cB1];
      }
      if (RELU) { v0 = fmaxf(v0, 0.f); v1 = fmaxf(v1, 0.f); }
      out[(size_t)row * N + cB0] = v0;
      out[(size_t)row * N + cB1] = v1;
    }
  }
}

// ---------------------------------------------------------------------------
// Per cross layer: fold weights.
//  Wqk[e, h*64+c] = sum_d wq[e,hd]*wk[c,hd]
//  bqk[h*64+c]    = sum_d bq[hd]  *wk[c,hd]        (bk drops out of softmax)
//  Wvo[h*64+c,d'] = sum_d wv[c,hd]*wo[hd,d']
//  bvo[d']        = sum_hd bv[hd]*wo[hd,d'] + bo[d']
// grid(256, 3), block 256.
// ---------------------------------------------------------------------------
__global__ void prep_cross_kernel(const float* __restrict__ wq,
                                  const float* __restrict__ bq,
                                  const float* __restrict__ wk,
                                  const float* __restrict__ wv,
                                  const float* __restrict__ bv,
                                  const float* __restrict__ wo,
                                  const float* __restrict__ bo,
                                  float* __restrict__ Wqk, float* __restrict__ bqk,
                                  float* __restrict__ Wvo, float* __restrict__ bvo) {
  const int t = threadIdx.x, x = blockIdx.x, y = blockIdx.y;
  if (y == 0) {
    const int h = t >> 6, c = t & 63;
    float a = 0.f;
    for (int d = 0; d < 64; ++d)
      a += wq[(size_t)x * DD + h * 64 + d] * wk[(size_t)c * DD + h * 64 + d];
    Wqk[(size_t)x * DD + t] = a;
  } else if (y == 1) {
    const int h = x >> 6, c = x & 63;
    float a = 0.f;
    for (int d = 0; d < 64; ++d)
      a += wv[(size_t)c * DD + h * 64 + d] * wo[(size_t)(h * 64 + d) * DD + t];
    Wvo[(size_t)x * DD + t] = a;
  } else {
    if (x == 0) {
      float a = bo[t];
      for (int e = 0; e < DD; ++e) a += bv[e] * wo[(size_t)e * DD + t];
      bvo[t] = a;
    } else if (x == 1) {
      const int h = t >> 6, c = t & 63;
      float a = 0.f;
      for (int d = 0; d < 64; ++d)
        a += bq[h * 64 + d] * wk[(size_t)c * DD + h * 64 + d];
      bqk[t] = a;
    }
  }
}

// ---------------------------------------------------------------------------
// Cross-attention core: one block per (b,i). Streams pair[b,i,:,:] (64 KB).
//  scores[h,j] = pair[j,:].qk[h,:] / 8 ; softmax over j ; wpair = w @ pair.
// This is the bandwidth-critical kernel (67 MB of pair per layer).
// ---------------------------------------------------------------------------
__global__ void cross_core_kernel(const float* __restrict__ pair,
                                  const float* __restrict__ qk,
                                  float* __restrict__ wpair) {
  __shared__ float qks[DD];
  __shared__ float wl[4 * LL];
  __shared__ float red[256];
  const int bi = blockIdx.x;       // b*L + i
  const int t  = threadIdx.x;      // j for pass 1
  qks[t] = qk[(size_t)bi * DD + t];
  __syncthreads();

  const float* pr = pair + (size_t)bi * LL * CBW + (size_t)t * CBW;
  __builtin_prefetch(pr, 0, 3);    // global_prefetch_b8 on the stream
  float s0 = 0.f, s1 = 0.f, s2 = 0.f, s3 = 0.f;
#pragma unroll 4
  for (int c = 0; c < CBW; ++c) {
    const float pv = pr[c];
    s0 += pv * qks[c];
    s1 += pv * qks[64 + c];
    s2 += pv * qks[128 + c];
    s3 += pv * qks[192 + c];
  }
  float sh[4] = {s0, s1, s2, s3};
  for (int h = 0; h < 4; ++h) {
    const float sv = sh[h] * 0.125f;             // 1/sqrt(hd=64)
    const float mx = block_reduce_max(sv, red);
    const float e  = __expf(sv - mx);
    const float sm = block_reduce_sum(e, red);
    wl[h * LL + t] = e / sm;
  }
  __syncthreads();

  // wpair[h*64+c] = sum_j w[h,j] * pair[j,c]   (thread -> (h,c))
  const int h = t >> 6, c = t & 63;
  const float* pb = pair + (size_t)bi * LL * CBW;
  float acc = 0.f;
#pragma unroll 8
  for (int j = 0; j < LL; ++j)
    acc += wl[h * LL + j] * pb[(size_t)j * CBW + c];
  wpair[(size_t)bi * DD + t] = acc;
}

// ---------------------------------------------------------------------------
// Row-wise LayerNorm in place: one block per row of 256.
// ---------------------------------------------------------------------------
__global__ void ln_kernel(float* __restrict__ y,
                          const float* __restrict__ g,
                          const float* __restrict__ b) {
  __shared__ float red[256];
  const int row = blockIdx.x, t = threadIdx.x;
  const float v = y[(size_t)row * DD + t];
  const float mean = block_reduce_sum(v, red) * (1.f / 256.f);
  const float d = v - mean;
  const float var = block_reduce_sum(d * d, red) * (1.f / 256.f);
  y[(size_t)row * DD + t] = d * rsqrtf(var + 1e-5f) * g[t] + b[t];
}

// ---------------------------------------------------------------------------
// Prepend CLS: xc[(b,0)] = cls ; xc[(b,t)] = h[(b,t-1)]. grid B*T, block 256.
// ---------------------------------------------------------------------------
__global__ void concat_cls_kernel(const float* __restrict__ cls,
                                  const float* __restrict__ h,
                                  float* __restrict__ xc) {
  const int row = blockIdx.x;
  const int b = row / TT, tt = row % TT, t = threadIdx.x;
  xc[(size_t)row * DD + t] =
      (tt == 0) ? cls[t] : h[(size_t)(b * LL + (tt - 1)) * DD + t];
}

// ---------------------------------------------------------------------------
// Self-attention core over T=257: one block per (b,q) row.
// ---------------------------------------------------------------------------
__global__ void self_core_kernel(const float* __restrict__ Qb,
                                 const float* __restrict__ Kb,
                                 const float* __restrict__ Vb,
                                 float* __restrict__ ctxb) {
  __shared__ float qs[DD];
  __shared__ float wl[4 * TT];
  __shared__ float red[256];
  const int gq = blockIdx.x;
  const int base = (gq / TT) * TT;          // row b*T
  const int t = threadIdx.x;
  qs[t] = Qb[(size_t)gq * DD + t];
  __syncthreads();

  float pm[4] = {-1e30f, -1e30f, -1e30f, -1e30f};
  for (int pass = 0; pass < 2; ++pass) {
    const int k = t + pass * 256;
    if (k < TT) {
      const float* kr = Kb + (size_t)(base + k) * DD;
      for (int h = 0; h < 4; ++h) {
        float s = 0.f;
#pragma unroll 8
        for (int d = 0; d < 64; ++d) s += qs[h * 64 + d] * kr[h * 64 + d];
        s *= 0.125f;
        wl[h * TT + k] = s;
        pm[h] = fmaxf(pm[h], s);
      }
    }
  }
  __syncthreads();
  float mx[4], sm[4];
  for (int h = 0; h < 4; ++h) mx[h] = block_reduce_max(pm[h], red);
  float ps[4] = {0.f, 0.f, 0.f, 0.f};
  for (int pass = 0; pass < 2; ++pass) {
    const int k = t + pass * 256;
    if (k < TT)
      for (int h = 0; h < 4; ++h) {
        const float e = __expf(wl[h * TT + k] - mx[h]);
        wl[h * TT + k] = e;
        ps[h] += e;
      }
  }
  for (int h = 0; h < 4; ++h) sm[h] = block_reduce_sum(ps[h], red);
  // reductions above include barriers -> wl visible block-wide here
  const int h = t >> 6, d = t & 63;
  float acc = 0.f;
  for (int k = 0; k < TT; ++k)
    acc += wl[h * TT + k] * Vb[(size_t)(base + k) * DD + h * 64 + d];
  ctxb[(size_t)gq * DD + t] = acc / sm[h];
}

// ---------------------------------------------------------------------------
// CLS-extract attention + head. One block per batch (b). All folded/scalar.
// (bk drops out of softmax, like in the cross layers.)
// ---------------------------------------------------------------------------
__global__ void finale_kernel(const float* __restrict__ hs,
                              const float* __restrict__ wq, const float* __restrict__ bq,
                              const float* __restrict__ wk,
                              const float* __restrict__ wv, const float* __restrict__ bv,
                              const float* __restrict__ wo, const float* __restrict__ bo,
                              const float* __restrict__ g,  const float* __restrict__ bb,
                              const float* __restrict__ hg, const float* __restrict__ hb,
                              const float* __restrict__ w1, const float* __restrict__ b1,
                              const float* __restrict__ w2, const float* __restrict__ b2,
                              float* __restrict__ out) {
  __shared__ float clsS[DD], Qs[DD], qk2[4 * DD], wl[4 * LL], wop[4 * DD];
  __shared__ float ctx[DD], red[256], f1[64];
  const int b = blockIdx.x, t = threadIdx.x;
  const float* cls = hs + (size_t)(b * TT) * DD;
  clsS[t] = cls[t];
  __syncthreads();

  // Q = cls @ wq + bq
  float a = bq[t];
  for (int e = 0; e < DD; ++e) a += clsS[e] * wq[(size_t)e * DD + t];
  Qs[t] = a;
  __syncthreads();

  // qk2[h][e] = sum_d Q[h,d] * wk[e, h*64+d]   (fold K-projection into Q)
  for (int h = 0; h < 4; ++h) {
    float s = 0.f;
    const float* wkr = wk + (size_t)t * DD + h * 64;
    for (int d = 0; d < 64; ++d) s += Qs[h * 64 + d] * wkr[d];
    qk2[h * DD + t] = s;
  }
  __syncthreads();

  // scores over the 256 non-CLS rows
  const float* othk = hs + (size_t)(b * TT + 1 + t) * DD;
  float s0 = 0.f, s1 = 0.f, s2 = 0.f, s3 = 0.f;
  for (int e = 0; e < DD; ++e) {
    const float ov = othk[e];
    s0 += ov * qk2[e];
    s1 += ov * qk2[DD + e];
    s2 += ov * qk2[2 * DD + e];
    s3 += ov * qk2[3 * DD + e];
  }
  float sh[4] = {s0, s1, s2, s3};
  for (int h = 0; h < 4; ++h) {
    const float sv = sh[h] * 0.125f;
    const float mx = block_reduce_max(sv, red);
    const float e  = __expf(sv - mx);
    const float su = block_reduce_sum(e, red);
    wl[h * LL + t] = e / su;
  }
  __syncthreads();

  // wop[h][e] = sum_k w[h,k] * oth[k][e]
  for (int h = 0; h < 4; ++h) {
    float s = 0.f;
    for (int k = 0; k < LL; ++k)
      s += wl[h * LL + k] * hs[(size_t)(b * TT + 1 + k) * DD + t];
    wop[h * DD + t] = s;
  }
  __syncthreads();

  // ctx[t=h*64+d] = sum_e wop[h,e]*wv[e,t] + bv[t]
  {
    const int h = t >> 6;
    float s = bv[t];
    for (int e = 0; e < DD; ++e) s += wop[h * DD + e] * wv[(size_t)e * DD + t];
    ctx[t] = s;
  }
  __syncthreads();

  // o = ctx @ wo + bo + cls ; LN(g,bb)
  float o = bo[t] + clsS[t];
  for (int e = 0; e < DD; ++e) o += ctx[e] * wo[(size_t)e * DD + t];
  float mean = block_reduce_sum(o, red) * (1.f / 256.f);
  float dv = o - mean;
  float var = block_reduce_sum(dv * dv, red) * (1.f / 256.f);
  const float cval = dv * rsqrtf(var + 1e-5f) * g[t] + bb[t];

  // head: LN(hg,hb) -> w1 relu -> w2
  mean = block_reduce_sum(cval, red) * (1.f / 256.f);
  dv = cval - mean;
  var = block_reduce_sum(dv * dv, red) * (1.f / 256.f);
  ctx[t] = dv * rsqrtf(var + 1e-5f) * hg[t] + hb[t];
  __syncthreads();
  if (t < 64) {
    float s = b1[t];
    for (int e = 0; e < DD; ++e) s += ctx[e] * w1[(size_t)e * 64 + t];
    f1[t] = s > 0.f ? s : 0.f;
  }
  __syncthreads();
  if (t == 0) {
    float s = b2[0];
    for (int j = 0; j < 64; ++j) s += f1[j] * w2[j];
    out[b] = s;
  }
}

// ---------------------------------------------------------------------------
// Host-side orchestration.
// Input order (dict insertion order): x, pair,
//  p_cross1{wq,bq,wk,bk,wv,bv,wo,bo,g,b}                      -> 2..11
//  p_cross2{...}                                              -> 12..21
//  p_self{cls,wq,bq,wk,bk,wv,bv,wo,bo,g1,b1n,w1,bf1,w2,bf2,g2,b2n} -> 22..38
//  p_ext{wq,bq,wk,bk,wv,bv,wo,bo,g,b}                         -> 39..48
//  p_head{g,b,w1,b1,w2,b2}                                    -> 49..54
// ---------------------------------------------------------------------------
extern "C" void kernel_launch(void* const* d_in, const int* in_sizes, int n_in,
                              void* d_out, int out_size, void* d_ws, size_t ws_size,
                              hipStream_t stream) {
  (void)in_sizes; (void)n_in; (void)out_size; (void)ws_size;
  const float* x    = (const float*)d_in[0];
  const float* pair = (const float*)d_in[1];
  auto in = [&](int i) { return (const float*)d_in[i]; };

  // Workspace layout (floats). Needs ~16.8 MB total.
  float* ws = (float*)d_ws;
  float* Wqk  = ws;                    // 65536
  float* Wvo  = Wqk  + 65536;          // 65536
  float* bqk  = Wvo  + 65536;          // 256
  float* bvo  = bqk  + 256;            // 256
  float* qk   = bvo  + 256;            // 1024*256
  float* wpr  = qk   + 262144;         // 1024*256
  float* buf0 = wpr  + 262144;         // 1024*256
  float* buf1 = buf0 + 262144;         // 1024*256
  float* xc   = buf1 + 262144;         // 1028*256
  float* Qb   = xc   + 263168;         // 1028*256
  float* Kb   = Qb   + 263168;         // 1028*256
  float* Vb   = Kb   + 263168;         // 1028*256
  float* ctxb = Vb   + 263168;         // 1028*256
  float* a0   = ctxb + 263168;         // 1028*256
  float* f1b  = a0   + 263168;         // 1028*1024
  float* f2b  = f1b  + 1052672;        // 1028*256
  _Float16* wpk = (_Float16*)(f2b + 263168); // 256*1024 f16 (512 KB)

  // Dispatch: pack weight to f16 fragment order, then GEMM.
  // Bias present for every call; variants cover (K, res, relu).
  auto gemm = [&](const float* A, const float* W, const float* bias,
                  const float* res, float* out, int M, int N, int K, int relu) {
    wpack_kernel<<<(K * N) / 256, 256, 0, stream>>>(W, wpk, N);
    dim3 grid(N / 32, (M + 127) / 128);
    if (K == 1024) {
      gemm_wmma_kernel<1024, true, false><<<grid, 256, 0, stream>>>(
          A, wpk, bias, res, out, M, N);
    } else if (res) {
      gemm_wmma_kernel<256, true, false><<<grid, 256, 0, stream>>>(
          A, wpk, bias, res, out, M, N);
    } else if (relu) {
      gemm_wmma_kernel<256, false, true><<<grid, 256, 0, stream>>>(
          A, wpk, bias, res, out, M, N);
    } else {
      gemm_wmma_kernel<256, false, false><<<grid, 256, 0, stream>>>(
          A, wpk, bias, res, out, M, N);
    }
  };

  // ---- cross layer 1 ----
  prep_cross_kernel<<<dim3(256, 3), 256, 0, stream>>>(
      in(2), in(3), in(4), in(6), in(7), in(8), in(9), Wqk, bqk, Wvo, bvo);
  gemm(x, Wqk, bqk, nullptr, qk, BB * LL, DD, DD, 0);
  cross_core_kernel<<<BB * LL, 256, 0, stream>>>(pair, qk, wpr);
  gemm(wpr, Wvo, bvo, x, buf0, BB * LL, DD, DD, 0);
  ln_kernel<<<BB * LL, 256, 0, stream>>>(buf0, in(10), in(11));

  // ---- cross layer 2 ----
  prep_cross_kernel<<<dim3(256, 3), 256, 0, stream>>>(
      in(12), in(13), in(14), in(16), in(17), in(18), in(19), Wqk, bqk, Wvo, bvo);
  gemm(buf0, Wqk, bqk, nullptr, qk, BB * LL, DD, DD, 0);
  cross_core_kernel<<<BB * LL, 256, 0, stream>>>(pair, qk, wpr);
  gemm(wpr, Wvo, bvo, buf0, buf1, BB * LL, DD, DD, 0);
  ln_kernel<<<BB * LL, 256, 0, stream>>>(buf1, in(20), in(21));

  // ---- self attention with CLS (T = 257) ----
  const int MT = BB * TT;  // 1028
  concat_cls_kernel<<<MT, 256, 0, stream>>>(in(22), buf1, xc);
  gemm(xc, in(23), in(24), nullptr, Qb, MT, DD, DD, 0);
  gemm(xc, in(25), in(26), nullptr, Kb, MT, DD, DD, 0);
  gemm(xc, in(27), in(28), nullptr, Vb, MT, DD, DD, 0);
  self_core_kernel<<<MT, 256, 0, stream>>>(Qb, Kb, Vb, ctxb);
  gemm(ctxb, in(29), in(30), xc, a0, MT, DD, DD, 0);
  ln_kernel<<<MT, 256, 0, stream>>>(a0, in(31), in(32));
  gemm(a0, in(33), in(34), nullptr, f1b, MT, 4 * DD, DD, 1);   // relu
  gemm(f1b, in(35), in(36), a0, f2b, MT, DD, 4 * DD, 0);
  ln_kernel<<<MT, 256, 0, stream>>>(f2b, in(37), in(38));

  // ---- CLS extract + head ----
  finale_kernel<<<BB, 256, 0, stream>>>(
      f2b,
      in(39), in(40), in(41), in(43), in(44), in(45), in(46), in(47), in(48),
      in(49), in(50), in(51), in(52), in(53), in(54),
      (float*)d_out);
}
